// GraphVAE_86844238725401
// MI455X (gfx1250) — compile-verified, matrix operand order checked by reading, not run on hardware
//
#include <hip/hip_runtime.h>

// ---------------------------------------------------------------------------
// GraphVAE forward for MI455X (gfx1250, wave32, WMMA).
// All GEMMs use v_wmma_f32_16x16x32_f16 (f16 inputs, f32 accumulate).
// Weights are pre-packed per call into the ISA 16-bit A/B fragment layout
// so the WMMA inner loop issues only contiguous b128 loads. Each wave
// register-blocks NB adjacent 16x16 N-tiles, reusing the A fragment.
// ---------------------------------------------------------------------------

typedef __attribute__((ext_vector_type(16))) _Float16 v16h;
typedef __attribute__((ext_vector_type(8)))  float    v8f;

#define N_NODES 524288
#define N_GRAPH 16384
#define N_EDGES 1048576
#define NODE_F  16
#define TRIU_C  741
#define NODE_C  608
#define EDGE_C  2812
#define HEAD_N  4161
#define HEAD_NP 4224   // padded to 264 n-tiles (divisible by NB=4)

union AFrag { uint4 q[2]; v16h v; };

// ---------------- degree / scatter helpers (GCN aggregation) ----------------

__global__ void k_deg_init(float* deg, int n) {
    int i = blockIdx.x * blockDim.x + threadIdx.x;
    if (i < n) deg[i] = 1.0f;               // self loop
}

__global__ void k_deg_accum(const long long* __restrict__ dst, float* deg, int e) {
    int i = blockIdx.x * blockDim.x + threadIdx.x;
    if (i < e) atomicAdd(&deg[(int)dst[i]], 1.0f);
}

__global__ void k_deg_rsqrt(float* deg, int n) {
    int i = blockIdx.x * blockDim.x + threadIdx.x;
    if (i < n) deg[i] = rsqrtf(deg[i]);
}

// agg[i][f] = dinv[i]^2 * t[i][f]   (self-loop contribution, also zero-init)
__global__ void k_agg_init(const float* __restrict__ t, const float* __restrict__ dinv,
                           float* __restrict__ agg, int n, int F) {
    long long g = (long long)blockIdx.x * blockDim.x + threadIdx.x;
    if (g >= (long long)n * F) return;
    int i = (int)(g / F);
    float d = dinv[i];
    agg[g] = d * d * t[g];
}

// agg[dst][f] += dinv[src]*dinv[dst] * t[src][f]
__global__ void k_edge_scatter(const float* __restrict__ t, const float* __restrict__ dinv,
                               const long long* __restrict__ src, const long long* __restrict__ dst,
                               float* __restrict__ agg, int e, int F) {
    long long g = (long long)blockIdx.x * blockDim.x + threadIdx.x;
    if (g >= (long long)e * F) return;
    int ei = (int)(g / F);
    int f  = (int)(g % F);
    int s = (int)src[ei], d = (int)dst[ei];
    float w = dinv[s] * dinv[d];
    atomicAdd(&agg[(size_t)d * F + f], w * t[(size_t)s * F + f]);
}

// h1h = f16(relu(agg + b))
__global__ void k_gcn_fin_relu_h(const float* __restrict__ agg, const float* __restrict__ b,
                                 _Float16* __restrict__ out, int n, int F) {
    long long g = (long long)blockIdx.x * blockDim.x + threadIdx.x;
    if (g >= (long long)n * F) return;
    int f = (int)(g % F);
    float v = agg[g] + b[f];
    out[g] = (_Float16)fmaxf(v, 0.0f);
}

// graph mean pool: gh[b][f] = f16( mean_{32 nodes}(agg2) + c2_b[f] )
__global__ void k_pool(const float* __restrict__ agg, const float* __restrict__ b,
                       _Float16* __restrict__ gh) {
    int g = blockIdx.x * blockDim.x + threadIdx.x;          // B*64
    if (g >= N_GRAPH * 64) return;
    int bi = g >> 6, f = g & 63;
    const float* p = agg + (size_t)bi * 32 * 64 + f;
    float s = 0.0f;
    #pragma unroll
    for (int i = 0; i < 32; ++i) s += p[(size_t)i * 64];
    gh[g] = (_Float16)(s * (1.0f / 32.0f) + b[f]);
}

// ---------------- f32 -> f16 convert with K padding ----------------

__global__ void k_cvt_pad(const float* __restrict__ in, _Float16* __restrict__ out,
                          int rows, int kin, int kpad) {
    long long g = (long long)blockIdx.x * blockDim.x + threadIdx.x;
    if (g >= (long long)rows * kpad) return;
    int r = (int)(g / kpad), c = (int)(g % kpad);
    out[g] = (c < kin) ? (_Float16)in[(size_t)r * kin + c] : (_Float16)0.0f;
}

// ---------------- weight packing into WMMA B-fragment layout ----------------
// Wp[((ktile*ntiles + tn)*32 + lane)*16 + j], with
//   k = ktile*32 + (j>>3)*16 + (lane>>4)*8 + (j&7),  n = tn*16 + (lane&15)

__global__ void k_pack_w(const float* __restrict__ W, _Float16* __restrict__ Wp,
                         int korig, int norig, int kpad, int npad) {
    int ntiles = npad >> 4;
    long long total = (long long)(kpad >> 5) * ntiles * 512;
    long long g = (long long)blockIdx.x * blockDim.x + threadIdx.x;
    if (g >= total) return;
    int j = (int)(g & 15);
    int lane = (int)((g >> 4) & 31);
    long long t = g >> 9;
    int tn = (int)(t % ntiles);
    int kt = (int)(t / ntiles);
    int k = kt * 32 + ((j >> 3) << 4) + ((lane >> 4) << 3) + (j & 7);
    int n = tn * 16 + (lane & 15);
    float v = (k < korig && n < norig) ? W[(size_t)k * norig + n] : 0.0f;
    Wp[g] = (_Float16)v;
}

// same, but columns are concat of three weight matrices (decoder heads)
__global__ void k_pack_w_head(const float* __restrict__ W0, const float* __restrict__ W1,
                              const float* __restrict__ W2, _Float16* __restrict__ Wp,
                              int K, int npad) {
    int ntiles = npad >> 4;
    long long total = (long long)(K >> 5) * ntiles * 512;
    long long g = (long long)blockIdx.x * blockDim.x + threadIdx.x;
    if (g >= total) return;
    int j = (int)(g & 15);
    int lane = (int)((g >> 4) & 31);
    long long t = g >> 9;
    int tn = (int)(t % ntiles);
    int kt = (int)(t / ntiles);
    int k = kt * 32 + ((j >> 3) << 4) + ((lane >> 4) << 3) + (j & 7);
    int n = tn * 16 + (lane & 15);
    float v = 0.0f;
    if (n < TRIU_C)                v = W0[(size_t)k * TRIU_C + n];
    else if (n < TRIU_C + NODE_C)  v = W1[(size_t)k * NODE_C + (n - TRIU_C)];
    else if (n < HEAD_N)           v = W2[(size_t)k * EDGE_C + (n - TRIU_C - NODE_C)];
    Wp[g] = (_Float16)v;
}

__global__ void k_pack_b_head(const float* __restrict__ b0, const float* __restrict__ b1,
                              const float* __restrict__ b2, float* __restrict__ bb) {
    int n = blockIdx.x * blockDim.x + threadIdx.x;
    if (n >= HEAD_NP) return;
    float v = 0.0f;
    if (n < TRIU_C)                v = b0[n];
    else if (n < TRIU_C + NODE_C)  v = b1[n - TRIU_C];
    else if (n < HEAD_N)           v = b2[n - TRIU_C - NODE_C];
    bb[n] = v;
}

// ---------------- WMMA GEMM: C = act(A @ W + bias) ----------------
// A: M x K f16 row-major (K % 32 == 0). Wp: packed. One wave computes a
// 16 x (16*NB) strip: A fragment loaded once per K-step, reused NB times.

template <int NB>
__global__ void __launch_bounds__(256)
k_gemm_wmma(const _Float16* __restrict__ A, const _Float16* __restrict__ Wp,
            const float* __restrict__ bias, float* __restrict__ Cf,
            _Float16* __restrict__ Ch, int M, int K, int npad, int relu) {
    int ntiles = npad >> 4;
    int gtn = ntiles / NB;                      // N-groups
    long long groups = (long long)(M >> 4) * gtn;
    long long grp = (long long)blockIdx.x * 8 + (threadIdx.x >> 5);
    if (grp >= groups) return;
    int tn0 = (int)(grp % gtn) * NB;
    int tm  = (int)(grp / gtn);
    int lane = threadIdx.x & 31;
    int hf = lane >> 4;
    int rc = lane & 15;

    const _Float16* arow = A + (size_t)(tm * 16 + rc) * K + hf * 8;
    const _Float16* bptr = Wp + ((size_t)tn0 * 32 + lane) * 16;
    size_t bstride = (size_t)ntiles * 512;      // halves per K-step

    v8f acc[NB];
    #pragma unroll
    for (int i = 0; i < NB; ++i) acc[i] = (v8f){};

    int kt = K >> 5;
    for (int k = 0; k < kt; ++k) {
        __builtin_prefetch(bptr + bstride, 0, 0);   // next K-step of packed W
        AFrag a;
        a.q[0] = *(const uint4*)(arow);
        a.q[1] = *(const uint4*)(arow + 16);
        AFrag b[NB];
        #pragma unroll
        for (int i = 0; i < NB; ++i) {
            b[i].q[0] = *(const uint4*)(bptr + (size_t)i * 512);
            b[i].q[1] = *(const uint4*)(bptr + (size_t)i * 512 + 8);
        }
        #pragma unroll
        for (int i = 0; i < NB; ++i)
            acc[i] = __builtin_amdgcn_wmma_f32_16x16x32_f16(false, a.v, false, b[i].v,
                                                            (short)0, acc[i], false, false);
        arow += 32;
        bptr += bstride;
    }

    #pragma unroll
    for (int i = 0; i < NB; ++i) {
        int n = (tn0 + i) * 16 + rc;
        float bv = bias ? bias[n] : 0.0f;
        #pragma unroll
        for (int r = 0; r < 8; ++r) {
            int m = tm * 16 + r + hf * 8;
            float v = acc[i][r] + bv;
            if (relu) v = fmaxf(v, 0.0f);
            size_t idx = (size_t)m * npad + n;
            if (Cf) Cf[idx] = v;
            if (Ch) Ch[idx] = (_Float16)v;
        }
    }
}

// ---------------- reparameterization & batchnorm ----------------

__global__ void k_reparam(const float* __restrict__ g2, const float* __restrict__ eps,
                          _Float16* __restrict__ zh) {
    int g = blockIdx.x * blockDim.x + threadIdx.x;          // B*128
    if (g >= N_GRAPH * 128) return;
    int b = g >> 7, f = g & 127;
    float mu = g2[(size_t)b * 256 + f];
    float ls = g2[(size_t)b * 256 + 128 + f];
    zh[g] = (_Float16)(eps[g] * __expf(ls) + mu);
}

__global__ void k_bn_stats(const float* __restrict__ P, int M, int F,
                           float* __restrict__ mean, float* __restrict__ rstd) {
    int f = blockIdx.x;
    __shared__ float sh[256], sh2[256];
    float s = 0.0f, s2 = 0.0f;
    for (int r = threadIdx.x; r < M; r += 256) {
        float v = P[(size_t)r * F + f];
        s += v; s2 += v * v;
    }
    sh[threadIdx.x] = s; sh2[threadIdx.x] = s2;
    __syncthreads();
    for (int o = 128; o > 0; o >>= 1) {
        if (threadIdx.x < o) { sh[threadIdx.x] += sh[threadIdx.x + o];
                               sh2[threadIdx.x] += sh2[threadIdx.x + o]; }
        __syncthreads();
    }
    if (threadIdx.x == 0) {
        float m = sh[0] / M;
        float var = sh2[0] / M - m * m;
        mean[f] = m;
        rstd[f] = rsqrtf(var + 1e-5f);
    }
}

__global__ void k_bn_apply_relu(const float* __restrict__ P, const float* __restrict__ gam,
                                const float* __restrict__ bet, const float* __restrict__ mean,
                                const float* __restrict__ rstd, _Float16* __restrict__ out,
                                int M, int F) {
    long long g = (long long)blockIdx.x * blockDim.x + threadIdx.x;
    if (g >= (long long)M * F) return;
    int f = (int)(g % F);
    float v = gam[f] * (P[g] - mean[f]) * rstd[f] + bet[f];
    out[g] = (_Float16)fmaxf(v, 0.0f);
}

// ---------------- split padded head output into the three results ----------

__global__ void k_split(const float* __restrict__ hc, float* __restrict__ out) {
    long long g = (long long)blockIdx.x * blockDim.x + threadIdx.x;
    if (g >= (long long)N_GRAPH * HEAD_N) return;
    int b = (int)(g / HEAD_N);
    int c = (int)(g % HEAD_N);
    float v = hc[(size_t)b * HEAD_NP + c];
    size_t o;
    if (c < TRIU_C)               o = (size_t)b * TRIU_C + c;
    else if (c < TRIU_C + NODE_C) o = (size_t)N_GRAPH * TRIU_C + (size_t)b * NODE_C + (c - TRIU_C);
    else                          o = (size_t)N_GRAPH * (TRIU_C + NODE_C) + (size_t)b * EDGE_C
                                      + (c - TRIU_C - NODE_C);
    out[o] = v;
}

// ---------------------------------------------------------------------------

static inline unsigned blocks_for(long long n, int t) { return (unsigned)((n + t - 1) / t); }
static inline unsigned gemm_blocks(long long m, int npad, int nb) {
    long long groups = (m >> 4) * ((npad >> 4) / nb);
    return (unsigned)((groups + 7) / 8);
}

extern "C" void kernel_launch(void* const* d_in, const int* in_sizes, int n_in,
                              void* d_out, int out_size, void* d_ws, size_t ws_size,
                              hipStream_t stream) {
    (void)in_sizes; (void)n_in; (void)out_size;
    const float*     x      = (const float*)d_in[0];
    const long long* eidx   = (const long long*)d_in[1];   // int64 in reference
    const float*     eps    = (const float*)d_in[3];
    const float *c1w = (const float*)d_in[4],  *c1b = (const float*)d_in[5];
    const float *c2w = (const float*)d_in[6],  *c2b = (const float*)d_in[7];
    const float *f1w = (const float*)d_in[8],  *f1b = (const float*)d_in[9];
    const float *f2w = (const float*)d_in[10], *f2b = (const float*)d_in[11];
    const float *d1w = (const float*)d_in[12], *d1b = (const float*)d_in[13];
    const float *d1g = (const float*)d_in[14], *d1t = (const float*)d_in[15];
    const float *d2w = (const float*)d_in[16], *d2b = (const float*)d_in[17];
    const float *d2g = (const float*)d_in[18], *d2t = (const float*)d_in[19];
    const float *d3w = (const float*)d_in[20], *d3b = (const float*)d_in[21];
    const float *d3g = (const float*)d_in[22], *d3t = (const float*)d_in[23];
    const float *aw  = (const float*)d_in[24], *ab  = (const float*)d_in[25];
    const float *nw  = (const float*)d_in[26], *nb  = (const float*)d_in[27];
    const float *ew  = (const float*)d_in[28], *eb  = (const float*)d_in[29];
    const long long* esrc = eidx;
    const long long* edst = eidx + N_EDGES;

    // ---- workspace layout (bump alloc; big GCN region reused by head GEMM)
    char* W = (char*)d_ws;
    size_t off = 0;
    auto alloc = [&](size_t bytes) { size_t p = off; off = (off + bytes + 255) & ~(size_t)255; return p; };
    const size_t NN = N_NODES, NG = N_GRAPH;
    size_t o_dinv  = alloc(NN * 4);
    size_t o_gh    = alloc(NG * 64 * 2);
    size_t o_g1h   = alloc(NG * 128 * 2);
    size_t o_g2    = alloc(NG * 256 * 4);
    size_t o_zh    = alloc(NG * 128 * 2);
    size_t o_b1    = alloc(NG * 128 * 4);
    size_t o_b1h   = alloc(NG * 128 * 2);
    size_t o_b2    = alloc(NG * 256 * 4);
    size_t o_b2h   = alloc(NG * 256 * 2);
    size_t o_b3    = alloc(NG * 512 * 4);
    size_t o_b3h   = alloc(NG * 512 * 2);
    size_t o_c1p   = alloc(32 * 32 * 2);
    size_t o_c2p   = alloc(32 * 64 * 2);
    size_t o_f1p   = alloc(64 * 128 * 2);
    size_t o_f2p   = alloc(128 * 256 * 2);
    size_t o_d1p   = alloc(128 * 128 * 2);
    size_t o_d2p   = alloc(128 * 256 * 2);
    size_t o_d3p   = alloc(256 * 512 * 2);
    size_t o_hp    = alloc((size_t)512 * HEAD_NP * 2);
    size_t o_hb    = alloc(HEAD_NP * 4);
    size_t o_mean  = alloc(512 * 4);
    size_t o_rstd  = alloc(512 * 4);
    size_t o_xh    = alloc(NN * 32 * 2);
    size_t big     = off;                         // reused region
    size_t o_t1    = big;                         // N x 32 f32
    size_t o_agg1  = o_t1   + ((NN * 32 * 4 + 255) & ~(size_t)255);
    size_t o_h1h   = o_agg1 + ((NN * 32 * 4 + 255) & ~(size_t)255);
    size_t o_t2    = o_h1h  + ((NN * 32 * 2 + 255) & ~(size_t)255);
    size_t o_agg2  = o_t2   + ((NN * 64 * 4 + 255) & ~(size_t)255);
    size_t end1    = o_agg2 + NN * 64 * 4;
    size_t o_headC = big;                         // B x HEAD_NP f32 (after pooling)
    size_t end2    = big + (size_t)NG * HEAD_NP * 4;
    size_t need    = (end1 > end2 ? end1 : end2);
    if (ws_size < need) return;                   // ~575 MB required

    float*     dinv = (float*)(W + o_dinv);
    _Float16*  xh   = (_Float16*)(W + o_xh);
    float*     t1   = (float*)(W + o_t1);
    float*     agg1 = (float*)(W + o_agg1);
    _Float16*  h1h  = (_Float16*)(W + o_h1h);
    float*     t2   = (float*)(W + o_t2);
    float*     agg2 = (float*)(W + o_agg2);
    _Float16*  gh   = (_Float16*)(W + o_gh);
    _Float16*  g1h  = (_Float16*)(W + o_g1h);
    float*     g2   = (float*)(W + o_g2);
    _Float16*  zh   = (_Float16*)(W + o_zh);
    float*     b1   = (float*)(W + o_b1);
    _Float16*  b1h  = (_Float16*)(W + o_b1h);
    float*     b2   = (float*)(W + o_b2);
    _Float16*  b2h  = (_Float16*)(W + o_b2h);
    float*     b3   = (float*)(W + o_b3);
    _Float16*  b3h  = (_Float16*)(W + o_b3h);
    float*     headC= (float*)(W + o_headC);
    float*     hbb  = (float*)(W + o_hb);
    float*     mean = (float*)(W + o_mean);
    float*     rstd = (float*)(W + o_rstd);

    const int T = 256;

    // ---- pack weights (tiny, every call -> deterministic)
    k_pack_w<<<blocks_for(1 * 2 * 512, T), T, 0, stream>>>(c1w, (_Float16*)(W + o_c1p), 16, 32, 32, 32);
    k_pack_w<<<blocks_for(1 * 4 * 512, T), T, 0, stream>>>(c2w, (_Float16*)(W + o_c2p), 32, 64, 32, 64);
    k_pack_w<<<blocks_for(2 * 8 * 512, T), T, 0, stream>>>(f1w, (_Float16*)(W + o_f1p), 64, 128, 64, 128);
    k_pack_w<<<blocks_for(4 * 16 * 512, T), T, 0, stream>>>(f2w, (_Float16*)(W + o_f2p), 128, 256, 128, 256);
    k_pack_w<<<blocks_for(4 * 8 * 512, T), T, 0, stream>>>(d1w, (_Float16*)(W + o_d1p), 128, 128, 128, 128);
    k_pack_w<<<blocks_for(4 * 16 * 512, T), T, 0, stream>>>(d2w, (_Float16*)(W + o_d2p), 128, 256, 128, 256);
    k_pack_w<<<blocks_for(8 * 32 * 512, T), T, 0, stream>>>(d3w, (_Float16*)(W + o_d3p), 256, 512, 256, 512);
    k_pack_w_head<<<blocks_for((long long)16 * (HEAD_NP / 16) * 512, T), T, 0, stream>>>(
        aw, nw, ew, (_Float16*)(W + o_hp), 512, HEAD_NP);
    k_pack_b_head<<<blocks_for(HEAD_NP, T), T, 0, stream>>>(ab, nb, eb, hbb);

    // ---- degree normalization (shared by both GCN layers)
    k_deg_init<<<blocks_for(N_NODES, T), T, 0, stream>>>(dinv, N_NODES);
    k_deg_accum<<<blocks_for(N_EDGES, T), T, 0, stream>>>(edst, dinv, N_EDGES);
    k_deg_rsqrt<<<blocks_for(N_NODES, T), T, 0, stream>>>(dinv, N_NODES);

    // ---- GCN layer 1: t1 = x @ c1_w (K padded 16->32), aggregate, relu
    k_cvt_pad<<<blocks_for((long long)N_NODES * 32, T), T, 0, stream>>>(x, xh, N_NODES, 16, 32);
    k_gemm_wmma<2><<<gemm_blocks(N_NODES, 32, 2), 256, 0, stream>>>(
        xh, (_Float16*)(W + o_c1p), nullptr, t1, nullptr, N_NODES, 32, 32, 0);
    k_agg_init<<<blocks_for((long long)N_NODES * 32, T), T, 0, stream>>>(t1, dinv, agg1, N_NODES, 32);
    k_edge_scatter<<<blocks_for((long long)N_EDGES * 32, T), T, 0, stream>>>(t1, dinv, esrc, edst, agg1, N_EDGES, 32);
    k_gcn_fin_relu_h<<<blocks_for((long long)N_NODES * 32, T), T, 0, stream>>>(agg1, c1b, h1h, N_NODES, 32);

    // ---- GCN layer 2: t2 = h1 @ c2_w, aggregate; bias folded into pooling
    k_gemm_wmma<4><<<gemm_blocks(N_NODES, 64, 4), 256, 0, stream>>>(
        h1h, (_Float16*)(W + o_c2p), nullptr, t2, nullptr, N_NODES, 32, 64, 0);
    k_agg_init<<<blocks_for((long long)N_NODES * 64, T), T, 0, stream>>>(t2, dinv, agg2, N_NODES, 64);
    k_edge_scatter<<<blocks_for((long long)N_EDGES * 64, T), T, 0, stream>>>(t2, dinv, esrc, edst, agg2, N_EDGES, 64);

    // ---- mean pool (32 nodes/graph) + c2_b
    k_pool<<<blocks_for((long long)N_GRAPH * 64, T), T, 0, stream>>>(agg2, c2b, gh);

    // ---- encoder MLP
    k_gemm_wmma<4><<<gemm_blocks(N_GRAPH, 128, 4), 256, 0, stream>>>(
        gh, (_Float16*)(W + o_f1p), f1b, nullptr, g1h, N_GRAPH, 64, 128, 1);
    k_gemm_wmma<4><<<gemm_blocks(N_GRAPH, 256, 4), 256, 0, stream>>>(
        g1h, (_Float16*)(W + o_f2p), f2b, g2, nullptr, N_GRAPH, 128, 256, 0);

    // ---- reparameterize
    k_reparam<<<blocks_for((long long)N_GRAPH * 128, T), T, 0, stream>>>(g2, eps, zh);

    // ---- decoder: GEMM -> BN -> relu (x3)
    k_gemm_wmma<4><<<gemm_blocks(N_GRAPH, 128, 4), 256, 0, stream>>>(
        zh, (_Float16*)(W + o_d1p), d1b, b1, nullptr, N_GRAPH, 128, 128, 0);
    k_bn_stats<<<128, 256, 0, stream>>>(b1, N_GRAPH, 128, mean, rstd);
    k_bn_apply_relu<<<blocks_for((long long)N_GRAPH * 128, T), T, 0, stream>>>(b1, d1g, d1t, mean, rstd, b1h, N_GRAPH, 128);

    k_gemm_wmma<4><<<gemm_blocks(N_GRAPH, 256, 4), 256, 0, stream>>>(
        b1h, (_Float16*)(W + o_d2p), d2b, b2, nullptr, N_GRAPH, 128, 256, 0);
    k_bn_stats<<<256, 256, 0, stream>>>(b2, N_GRAPH, 256, mean, rstd);
    k_bn_apply_relu<<<blocks_for((long long)N_GRAPH * 256, T), T, 0, stream>>>(b2, d2g, d2t, mean, rstd, b2h, N_GRAPH, 256);

    k_gemm_wmma<4><<<gemm_blocks(N_GRAPH, 512, 4), 256, 0, stream>>>(
        b2h, (_Float16*)(W + o_d3p), d3b, b3, nullptr, N_GRAPH, 256, 512, 0);
    k_bn_stats<<<512, 256, 0, stream>>>(b3, N_GRAPH, 512, mean, rstd);
    k_bn_apply_relu<<<blocks_for((long long)N_GRAPH * 512, T), T, 0, stream>>>(b3, d3g, d3t, mean, rstd, b3h, N_GRAPH, 512);

    // ---- fused decoder heads (512 -> 4224 padded), then split into d_out
    k_gemm_wmma<4><<<gemm_blocks(N_GRAPH, HEAD_NP, 4), 256, 0, stream>>>(
        b3h, (_Float16*)(W + o_hp), hbb, headC, nullptr, N_GRAPH, 512, HEAD_NP, 0);
    k_split<<<blocks_for((long long)N_GRAPH * HEAD_N, T), T, 0, stream>>>(headC, (float*)d_out);
}
